// CosVQ_Reactivation_1657857376705
// MI455X (gfx1250) — matrix operand order repaired and checked by reading, not compiled, not run on hardware
//
#include <hip/hip_runtime.h>
#include <hip/hip_bf16.h>
#include <stdint.h>

// ---------------------------------------------------------------------------
// CosVQ reactivation forward, MI455X (gfx1250, wave32, WMMA).
//
//   N=16384 tokens, D=128, K=8192 codes.
//   GEMM 2*N*K*D = 34.4 GFLOP done twice via v_wmma_f32_16x16x32_f16.
//   HBM traffic ~30MB -> compute-bound; f16 codebook (2MB) is L2 resident.
//   cos<=1  =>  logits/T <= 10: fixed softmax max (no online rescale).
//   B fragments double-buffered in VGPRs so WMMA only sees partial waits.
//   exp(s/T - 10) computed as exp2(fma(s, 10*log2e, -10*log2e)): one FMA +
//   one v_exp_f32 per score (v_exp_f32 is base-2; avoid the extra pk_mul).
// ---------------------------------------------------------------------------

typedef __attribute__((ext_vector_type(16))) _Float16 v16h;
typedef __attribute__((ext_vector_type(8)))  _Float16 v8h;
typedef __attribute__((ext_vector_type(4)))  _Float16 v4h;
typedef __attribute__((ext_vector_type(8)))  float    v8f;

#define DDIM     128
#define BETA     0.25f
#define EPS_LOG  1e-8f
#define KMAX     8192            // LDS colsum capacity (32KB of 320KB/WGP)
#define L2E10    14.4269504089f  // (1/T)*log2(e) = 10*log2(e); SMAX folds to -L2E10

// ---------------------------------------------------------------------------
// Row L2-normalize f32 -> f16, one wave per row (D=128 -> 4 floats/lane).
// ---------------------------------------------------------------------------
__global__ void vq_normalize_f16(const float* __restrict__ x,
                                 _Float16* __restrict__ y, int rows) {
  int row  = blockIdx.x * (blockDim.x >> 5) + (threadIdx.x >> 5);
  int lane = threadIdx.x & 31;
  if (row >= rows) return;
  const float4 v = ((const float4*)(x + (size_t)row * DDIM))[lane];
  float ss = v.x * v.x + v.y * v.y + v.z * v.z + v.w * v.w;
#pragma unroll
  for (int off = 16; off > 0; off >>= 1) ss += __shfl_xor(ss, off, 32);
  float inv = 1.0f / fmaxf(sqrtf(ss), 1e-12f);   // torch F.normalize eps
  v4h o;
  o[0] = (_Float16)(v.x * inv); o[1] = (_Float16)(v.y * inv);
  o[2] = (_Float16)(v.z * inv); o[3] = (_Float16)(v.w * inv);
  *(v4h*)(y + (size_t)row * DDIM + lane * 4) = o;
}

// ---------------------------------------------------------------------------
// WMMA fragments as loaded here (wave32, f16 16x16x32); same k order for A and B.
// A 16x32 (MxK): lane = M row; a[0..15]=K kh..kh+15 contiguous (kh=16*(lane>>4))
// B 32x16 (KxN): lane = N col; b[0..15]=K kh..kh+15 contiguous (kh=16*(lane>>4))
// ---------------------------------------------------------------------------
__device__ __forceinline__ v16h load_a_frag(const _Float16* __restrict__ base,
                                            int row0, int d0, int lane) {
  const _Float16* p = base + (size_t)(row0 + (lane & 15)) * DDIM + d0 + ((lane >> 4) << 4);
  v8h lo = *(const v8h*)(p);
  v8h hi = *(const v8h*)(p + 8);
  v16h f;
#pragma unroll
  for (int i = 0; i < 8; ++i) { f[i] = lo[i]; f[8 + i] = hi[i]; }
  return f;
}

// Four B fragments (D=128) of one 16-col tile; q = pb + kt*DDIM.
// Immediate-offset b128 loads from a single per-lane pointer.
__device__ __forceinline__ void load_b4(const _Float16* __restrict__ q,
                                        v16h& b0, v16h& b1, v16h& b2, v16h& b3) {
  b0 = *(const v16h*)(q);
  b1 = *(const v16h*)(q + 32);
  b2 = *(const v16h*)(q + 64);
  b3 = *(const v16h*)(q + 96);
}

__device__ __forceinline__ v8f wmma4(v16h a0, v16h a1, v16h a2, v16h a3,
                                     v16h b0, v16h b1, v16h b2, v16h b3) {
  v8f c = {};
  c = __builtin_amdgcn_wmma_f32_16x16x32_f16(false, a0, false, b0, (short)0, c, false, false);
  c = __builtin_amdgcn_wmma_f32_16x16x32_f16(false, a1, false, b1, (short)0, c, false, false);
  c = __builtin_amdgcn_wmma_f32_16x16x32_f16(false, a2, false, b2, (short)0, c, false, false);
  c = __builtin_amdgcn_wmma_f32_16x16x32_f16(false, a3, false, b3, (short)0, c, false, false);
  return c;
}

// phase 1 per-tile: running sum-exp (fixed max) + running argmax
__device__ __forceinline__ void p1_process(const v8f& c, int kcol,
                                           float* zsum, float* bv, int* bi) {
#pragma unroll
  for (int j = 0; j < 8; ++j) {
    float s  = c[j];
    bool  up = s > bv[j];                  // strict > keeps earliest (cols ascend)
    bv[j] = up ? s    : bv[j];
    bi[j] = up ? kcol : bi[j];
    zsum[j] += exp2f(__builtin_fmaf(s, L2E10, -L2E10));
  }
}

// phase 2 per-tile: softmax prob column-sum into LDS
__device__ __forceinline__ void p2_process(const v8f& c, int kt, int col, int lane,
                                           const float* rz, float* lds_colsum) {
  float acc = 0.0f;
#pragma unroll
  for (int j = 0; j < 8; ++j)
    acc += exp2f(__builtin_fmaf(c[j], L2E10, -L2E10)) * rz[j];
  acc += __shfl_xor(acc, 16, 32);          // fold rows 8-15 half onto 0-7 half
  if (lane < 16) atomicAdd(&lds_colsum[kt + col], acc);
}

// ---------------------------------------------------------------------------
// Main kernel: 256 threads = 8 waves, 16 rows/wave => 128 rows/block.
// Phase 1: sweep K in 16-col WMMA tiles -> per-row sum-exp Z and argmax.
// Phase 2: re-sweep (codebook from L2), accumulate softmax column sums in LDS.
// K-loop unrolled x2 with two B register buffers: loads run one tile ahead.
// C/D layout: lane(0..15) holds col kt+lane, rows 0..7 in c[0..7];
//             lane(16..31) holds col kt+lane-16, rows 8..15.
// Requires K % 32 == 0 (K = 8192 here).
// ---------------------------------------------------------------------------
__global__ __launch_bounds__(256)
void vq_score_kernel(const _Float16* __restrict__ zn,
                     const _Float16* __restrict__ cbn,
                     int*   __restrict__ idx_out,
                     float* __restrict__ colsum_g,
                     int K) {
  __shared__ float lds_colsum[KMAX];
  for (int k = threadIdx.x; k < K; k += blockDim.x) lds_colsum[k] = 0.0f;
  __syncthreads();

  const int lane    = threadIdx.x & 31;
  const int wave    = threadIdx.x >> 5;
  const int rowBase = blockIdx.x * 128 + wave * 16;
  const int col     = lane & 15;

  // A fragments for this wave's 16 rows, all of D=128 (live across both phases)
  const v16h a0 = load_a_frag(zn, rowBase,  0, lane);
  const v16h a1 = load_a_frag(zn, rowBase, 32, lane);
  const v16h a2 = load_a_frag(zn, rowBase, 64, lane);
  const v16h a3 = load_a_frag(zn, rowBase, 96, lane);

  // per-lane B pointer: fragment = 4 x 32B at byte offsets {0,64,128,192} of
  // pb + kt*256B  (row stride 128 f16 = 256B)
  const _Float16* pb = cbn + (size_t)col * DDIM + ((lane >> 4) << 4);

  float zsum[8], bv[8];
  int   bi[8];
#pragma unroll
  for (int j = 0; j < 8; ++j) { zsum[j] = 0.0f; bv[j] = -3.0e38f; bi[j] = 0; }

  v16h b0, b1, b2, b3, e0, e1, e2, e3;

  // ---- phase 1: Z (sum-exp at fixed max) + argmax -------------------------
  load_b4(pb, b0, b1, b2, b3);
  for (int kt = 0; kt < K; kt += 32) {
    load_b4(pb + (size_t)(kt + 16) * DDIM, e0, e1, e2, e3);
    p1_process(wmma4(a0, a1, a2, a3, b0, b1, b2, b3), kt + col, zsum, bv, bi);
    int kw = (kt + 32 < K) ? kt + 32 : 0;
    load_b4(pb + (size_t)kw * DDIM, b0, b1, b2, b3);
    p1_process(wmma4(a0, a1, a2, a3, e0, e1, e2, e3), kt + 16 + col, zsum, bv, bi);
  }

  // cross-lane combine within each 16-lane half (rows 0-7 / 8-15)
#pragma unroll
  for (int off = 1; off < 16; off <<= 1) {
#pragma unroll
    for (int j = 0; j < 8; ++j) {
      float zo  = __shfl_xor(zsum[j], off, 32);
      float bvo = __shfl_xor(bv[j],   off, 32);
      int   bio = __shfl_xor(bi[j],   off, 32);
      zsum[j] += zo;
      bool take = (bvo > bv[j]) || (bvo == bv[j] && bio < bi[j]);
      bv[j] = take ? bvo : bv[j];
      bi[j] = take ? bio : bi[j];
    }
  }
  if (lane == 0 || lane == 16) {
    int r0 = rowBase + ((lane >> 4) << 3);
#pragma unroll
    for (int j = 0; j < 8; ++j) idx_out[r0 + j] = bi[j];
  }
  float rz[8];
#pragma unroll
  for (int j = 0; j < 8; ++j) rz[j] = 1.0f / zsum[j];

  // ---- phase 2: accumulate P column sums (codebook re-read hits L2) -------
  load_b4(pb, b0, b1, b2, b3);
  for (int kt = 0; kt < K; kt += 32) {
    load_b4(pb + (size_t)(kt + 16) * DDIM, e0, e1, e2, e3);
    p2_process(wmma4(a0, a1, a2, a3, b0, b1, b2, b3), kt, col, lane, rz, lds_colsum);
    int kw = (kt + 32 < K) ? kt + 32 : 0;
    load_b4(pb + (size_t)kw * DDIM, b0, b1, b2, b3);
    p2_process(wmma4(a0, a1, a2, a3, e0, e1, e2, e3), kt + 16, col, lane, rz, lds_colsum);
  }
  __syncthreads();
  for (int k = threadIdx.x; k < K; k += blockDim.x)
    atomicAdd(&colsum_g[k], lds_colsum[k]);
}

// ---------------------------------------------------------------------------
// Gather z_q = codebook[idx] (== z_q_ste numerically), commit SSE, histogram.
// One wave per row.
// ---------------------------------------------------------------------------
__global__ void vq_gather_kernel(const float* __restrict__ z,
                                 const float* __restrict__ cb,
                                 const int*   __restrict__ idx,
                                 float* __restrict__ zq,
                                 int*   __restrict__ counts,
                                 float* __restrict__ sumsq,
                                 int N) {
  int row  = blockIdx.x * (blockDim.x >> 5) + (threadIdx.x >> 5);
  int lane = threadIdx.x & 31;
  if (row >= N) return;
  int id = idx[row];
  float4 cv = ((const float4*)(cb + (size_t)id  * DDIM))[lane];
  float4 zv = ((const float4*)(z  + (size_t)row * DDIM))[lane];
  ((float4*)(zq + (size_t)row * DDIM))[lane] = cv;
  float dx = cv.x - zv.x, dy = cv.y - zv.y, dz = cv.z - zv.z, dw = cv.w - zv.w;
  float d = dx * dx + dy * dy + dz * dz + dw * dw;
#pragma unroll
  for (int off = 16; off > 0; off >>= 1) d += __shfl_xor(d, off, 32);
  if (lane == 0) {
    atomicAdd(sumsq, d);
    atomicAdd(&counts[id], 1);
  }
}

// ---------------------------------------------------------------------------
// Scalars: commit loss, perplexity, entropy loss, min(ema).
// K <= N here, so "rank among dead < N" always holds -> sel == dead.
// ---------------------------------------------------------------------------
__global__ __launch_bounds__(256)
void vq_finalize_kernel(const int*   __restrict__ counts,
                        const float* __restrict__ colsum,
                        const float* __restrict__ sumsq,
                        const float* __restrict__ probs_ema,
                        float* __restrict__ out4, int N, int K) {
  __shared__ float red[256];
  const int tid = threadIdx.x;
  const float invN   = 1.0f / (float)N;
  const float thresh = 0.0125f / (float)K;
  const float fillv  = 1.0f / (float)K;
  float sp = 0.0f, se = 0.0f, mn = 3.4e38f;
  for (int k = tid; k < K; k += 256) {
    float em = (float)counts[k] * invN;
    sp += em * __logf(em + EPS_LOG);
    float pa = colsum[k] * invN + EPS_LOG;
    se += pa * __logf(pa);
    float ema = 0.9f * probs_ema[k] + 0.1f * em;
    if (ema < thresh) ema = fillv;
    mn = fminf(mn, ema);
  }
  red[tid] = sp; __syncthreads();
  for (int s = 128; s > 0; s >>= 1) { if (tid < s) red[tid] += red[tid + s]; __syncthreads(); }
  float sp_t = red[0]; __syncthreads();
  red[tid] = se; __syncthreads();
  for (int s = 128; s > 0; s >>= 1) { if (tid < s) red[tid] += red[tid + s]; __syncthreads(); }
  float se_t = red[0]; __syncthreads();
  red[tid] = mn; __syncthreads();
  for (int s = 128; s > 0; s >>= 1) { if (tid < s) red[tid] = fminf(red[tid], red[tid + s]); __syncthreads(); }
  if (tid == 0) {
    out4[0] = (1.0f + BETA) * sumsq[0] / ((float)N * (float)DDIM);  // commit
    out4[1] = __expf(-sp_t);                                        // perplexity
    out4[2] = -se_t;                                                // entropy loss
    out4[3] = red[0];                                               // min(ema)
  }
}

__global__ void vq_zero_kernel(int* __restrict__ counts,
                               float* __restrict__ colsum,
                               float* __restrict__ sumsq, int K) {
  int k = blockIdx.x * blockDim.x + threadIdx.x;
  if (k < K) { counts[k] = 0; colsum[k] = 0.0f; }
  if (k == 0) sumsq[0] = 0.0f;
}

// ---------------------------------------------------------------------------
extern "C" void kernel_launch(void* const* d_in, const int* in_sizes, int n_in,
                              void* d_out, int out_size, void* d_ws, size_t ws_size,
                              hipStream_t stream) {
  const float* z  = (const float*)d_in[0];   // [N, 128]
  const float* cb = (const float*)d_in[1];   // [K, 128]
  const float* pe = (const float*)d_in[2];   // [K]
  const int N = in_sizes[0] / DDIM;          // 16384
  const int K = in_sizes[1] / DDIM;          // 8192

  // workspace layout (~6.5 MB): zn, cbn (f16), idx, counts, colsum, sumsq
  auto al = [](uintptr_t x) { return (x + 255) & ~(uintptr_t)255; };
  uintptr_t p = (uintptr_t)d_ws;
  _Float16* zn     = (_Float16*)p; p = al(p + (size_t)N * DDIM * 2);
  _Float16* cbn    = (_Float16*)p; p = al(p + (size_t)K * DDIM * 2);
  int*      idx    = (int*)p;      p = al(p + (size_t)N * 4);
  int*      counts = (int*)p;      p = al(p + (size_t)K * 4);
  float*    colsum = (float*)p;    p = al(p + (size_t)K * 4);
  float*    sumsq  = (float*)p;

  float* zq   = (float*)d_out;
  float* out4 = (float*)d_out + (out_size - 4);

  vq_zero_kernel  <<<(K + 255) / 256, 256, 0, stream>>>(counts, colsum, sumsq, K);
  vq_normalize_f16<<<(N + 7) / 8,     256, 0, stream>>>(z,  zn,  N);
  vq_normalize_f16<<<(K + 7) / 8,     256, 0, stream>>>(cb, cbn, K);
  vq_score_kernel <<<N / 128,         256, 0, stream>>>(zn, cbn, idx, colsum, K);
  vq_gather_kernel<<<(N + 7) / 8,     256, 0, stream>>>(z, cb, idx, zq, counts, sumsq, N);
  vq_finalize_kernel<<<1,             256, 0, stream>>>(counts, colsum, sumsq, pe, out4, N, K);
}
